// InCTRLPQA_2980707303619
// MI455X (gfx1250) — matrix operand order repaired and checked by reading, not compiled, not run on hardware
//
#include <hip/hip_runtime.h>
#include <hip/hip_bf16.h>

typedef __attribute__((ext_vector_type(16))) __bf16 v16bf;
typedef __attribute__((ext_vector_type(8)))  __bf16 v8bf;
typedef __attribute__((ext_vector_type(8)))  float  v8f;

#define L_   3
#define B_   16
#define N_   256
#define D_   768
#define M_   2048
#define H_   128
#define MID_ 64
#define G_   16
#define IMG_ 240

// ---------------------------------------------------------------------------
// WMMA helpers (CDNA5 gfx1250, wave32). Contiguous k order used here (same for A and B):
//  A 16x32 bf16: lane<16 -> row=lane, K {0..15};       lane>=16 -> K {16..31}
//  B 32x16 bf16: lane<16 -> col=lane, K {0..15};       lane>=16 -> K {16..31}
// ---------------------------------------------------------------------------
__device__ __forceinline__ v8f wmma_bf16(v16bf a, v16bf b, v8f c) {
  return __builtin_amdgcn_wmma_f32_16x16x32_bf16(false, a, false, b, (short)0, c,
                                                 false, false);
}

__device__ __forceinline__ v8f zero8f() {
  v8f z;
#pragma unroll
  for (int i = 0; i < 8; ++i) z[i] = 0.f;
  return z;
}

__device__ __forceinline__ v16bf zero16bf() {
  v16bf z;
#pragma unroll
  for (int i = 0; i < 16; ++i) z[i] = (__bf16)0.0f;
  return z;
}

__device__ __forceinline__ v16bf frag_a_bf16(const __bf16* base, int ld, int lane) {
  int row = lane & 15;
  int off = (lane >> 4) ? 16 : 0;
  const __bf16* p = base + (size_t)row * ld + off;
  v8bf lo = *(const v8bf*)p;
  v8bf hi = *(const v8bf*)(p + 8);
  v16bf f;
#pragma unroll
  for (int i = 0; i < 8; ++i) { f[i] = lo[i]; f[8 + i] = hi[i]; }
  return f;
}

__device__ __forceinline__ v16bf frag_a_f32(const float* base, int ld, int lane) {
  int row = lane & 15;
  int off = (lane >> 4) ? 16 : 0;
  const float* p = base + (size_t)row * ld + off;
  v16bf f;
#pragma unroll
  for (int i = 0; i < 8; ++i) f[i] = (__bf16)p[i];
#pragma unroll
  for (int i = 0; i < 8; ++i) f[8 + i] = (__bf16)p[8 + i];
  return f;
}

__device__ __forceinline__ v16bf frag_b_f32_scaled(const float* base, int ld, int lane,
                                                   float s) {
  int col = lane & 15;
  int off = (lane >> 4) ? 16 : 0;
  const float* p = base + (size_t)col * ld + off;
  v16bf f;
#pragma unroll
  for (int i = 0; i < 16; ++i) f[i] = (__bf16)(p[i] * s);
  return f;
}

// ---------------------------------------------------------------------------
// 1) row-wise L2 normalization (768 cols). dstN and/or dstInv optional.
// ---------------------------------------------------------------------------
__global__ void rownorm_kernel(const float* __restrict__ src, float* __restrict__ dstN,
                               float* __restrict__ dstInv, int rows) {
  __shared__ float sm[256];
  int row = blockIdx.x;
  if (row >= rows) return;
  const float* p = src + (size_t)row * D_;
  int t = threadIdx.x;
  float s = 0.f;
  for (int i = t; i < D_; i += 256) { float v = p[i]; s += v * v; }
  sm[t] = s; __syncthreads();
  for (int k = 128; k > 0; k >>= 1) { if (t < k) sm[t] += sm[t + k]; __syncthreads(); }
  float inv = 1.0f / fmaxf(sqrtf(sm[0]), 1e-12f);
  if (dstInv && t == 0) dstInv[row] = inv;
  if (dstN) {
    float* o = dstN + (size_t)row * D_;
    for (int i = t; i < D_; i += 256) o[i] = p[i] * inv;
  }
}

// ---------------------------------------------------------------------------
// 2) cosine-sim GEMM [16 x 2048] per (l,b,nblock) with fused rowwise max/argmax.
//    A = normalized query rows (f32->bf16), B = raw prompt rows scaled by 1/||row||.
// ---------------------------------------------------------------------------
__global__ void sim_kernel(const float* __restrict__ qn, const float* __restrict__ prompt,
                           const float* __restrict__ prnorm, float* __restrict__ maxcos,
                           int* __restrict__ idxbuf) {
  __shared__ float T[8 * 256];
  __shared__ float redV[8][16];
  __shared__ int   redI[8][16];
  int blk = blockIdx.x;
  int nb = blk & 15;
  int lb = blk >> 4;  // l*16+b
  int lane = threadIdx.x & 31, wid = threadIdx.x >> 5;
  const float* qbase = qn + ((size_t)lb * N_ + nb * 16) * D_;
  const float* pbase = prompt + (size_t)lb * M_ * D_;
  const float* rs = prnorm + (size_t)lb * M_;
  float bestv = -1e30f;
  int besti = 0;
  float* Tw = T + wid * 256;
  int nloc = lane & 15;
  for (int mt = wid; mt < 128; mt += 8) {
    int m0 = mt * 16;
    float s = rs[m0 + nloc];
    const float* pb = pbase + (size_t)m0 * D_;
    if (mt + 8 < 128) __builtin_prefetch(pb + (size_t)128 * D_, 0, 1);
    v8f acc = zero8f();
    for (int k0 = 0; k0 < D_; k0 += 32) {
      v16bf a = frag_a_f32(qbase + k0, D_, lane);
      v16bf bb = frag_b_f32_scaled(pb + k0, D_, lane, s);
      acc = wmma_bf16(a, bb, acc);
    }
    int mh = (lane >> 4) * 8;
#pragma unroll
    for (int r = 0; r < 8; ++r) Tw[(mh + r) * 16 + nloc] = acc[r];
    __syncthreads();
    if (lane < 16) {
#pragma unroll
      for (int j = 0; j < 16; ++j) {
        float v = Tw[lane * 16 + j];
        if (v > bestv) { bestv = v; besti = m0 + j; }
      }
    }
    __syncthreads();
  }
  if (lane < 16) { redV[wid][lane] = bestv; redI[wid][lane] = besti; }
  __syncthreads();
  if (threadIdx.x < 16) {
    float bv = redV[0][threadIdx.x];
    int bi = redI[0][threadIdx.x];
    for (int w = 1; w < 8; ++w) {
      float v = redV[w][threadIdx.x];
      int mi = redI[w][threadIdx.x];
      if (v > bv || (v == bv && mi < bi)) { bv = v; bi = mi; }
    }
    int row = lb * N_ + nb * 16 + threadIdx.x;
    maxcos[row] = bv;
    idxbuf[row] = bi;
  }
}

// ---------------------------------------------------------------------------
// 3) gather aligned prompt rows, ctx = qn + |qn - l2norm(aligned)|, residual, idx.
// ---------------------------------------------------------------------------
__global__ void gather_ctx_kernel(const float* __restrict__ prompt,
                                  const float* __restrict__ qn,
                                  const float* __restrict__ maxcos,
                                  const int* __restrict__ idxbuf, float* __restrict__ ctx,
                                  float* __restrict__ out, size_t O_RES, size_t O_IDX,
                                  size_t O_AL) {
  __shared__ float sm[256];
  int blk = blockIdx.x;
  int n = blk & 255;
  int lb = blk >> 8;
  int row = lb * N_ + n;
  int t = threadIdx.x;
  int idx = idxbuf[row];
  const float* arow = prompt + ((size_t)lb * M_ + idx) * D_;
  float s = 0.f;
  for (int i = t; i < D_; i += 256) { float v = arow[i]; s += v * v; }
  sm[t] = s; __syncthreads();
  for (int k = 128; k > 0; k >>= 1) { if (t < k) sm[t] += sm[t + k]; __syncthreads(); }
  float inv = 1.0f / fmaxf(sqrtf(sm[0]), 1e-12f);
  const float* qrow = qn + (size_t)row * D_;
  float* crow = ctx + (size_t)row * D_;
  float* alrow = out + O_AL + (size_t)row * D_;
  for (int i = t; i < D_; i += 256) {
    float a = arow[i];
    alrow[i] = a;
    float an = a * inv;
    float q = qrow[i];
    crow[i] = q + fabsf(q - an);
  }
  if (t == 0) {
    out[O_RES + row] = 0.5f * (1.0f - maxcos[row]);
    out[O_IDX + row] = (float)idx;
  }
}

// ---------------------------------------------------------------------------
// 4) BatchNorm stats for ctx (per (l,d) over B*N, biased var) -> scale/shift.
// ---------------------------------------------------------------------------
__global__ void bnstats_ctx_kernel(const float* __restrict__ ctx,
                                   const float* __restrict__ g,
                                   const float* __restrict__ be, float* __restrict__ scale,
                                   float* __restrict__ shift) {
  __shared__ float s1[256], s2[256];
  int d = blockIdx.x % D_;
  int l = blockIdx.x / D_;
  int t = threadIdx.x;
  float a = 0.f, b2 = 0.f;
  for (int i = t; i < B_ * N_; i += 256) {
    float v = ctx[((size_t)l * B_ * N_ + i) * D_ + d];
    a += v; b2 += v * v;
  }
  s1[t] = a; s2[t] = b2; __syncthreads();
  for (int k = 128; k > 0; k >>= 1) {
    if (t < k) { s1[t] += s1[t + k]; s2[t] += s2[t + k]; }
    __syncthreads();
  }
  if (t == 0) {
    float mean = s1[0] * (1.0f / (B_ * N_));
    float var = s2[0] * (1.0f / (B_ * N_)) - mean * mean;
    float sc = g[l * D_ + d] * rsqrtf(var + 1e-5f);
    scale[l * D_ + d] = sc;
    shift[l * D_ + d] = be[l * D_ + d] - mean * sc;
  }
}

// ---------------------------------------------------------------------------
// 5) Apply BN to ctx -> bf16 spatial-major [lb][n][d]; fused GAP/GMP per (lb,d).
// ---------------------------------------------------------------------------
__global__ void bnapply_ctx_kernel(const float* __restrict__ ctx,
                                   const float* __restrict__ scale,
                                   const float* __restrict__ shift,
                                   __bf16* __restrict__ ctxbf, float* __restrict__ gap,
                                   float* __restrict__ gmp) {
  int lb = blockIdx.x;
  int l = lb >> 4;
  int t = threadIdx.x;
  float sum[3] = {0.f, 0.f, 0.f};
  float mx[3] = {-1e30f, -1e30f, -1e30f};
  float sc[3], sh[3];
#pragma unroll
  for (int j = 0; j < 3; ++j) {
    int d = t + j * 256;
    sc[j] = scale[l * D_ + d];
    sh[j] = shift[l * D_ + d];
  }
  for (int n = 0; n < N_; ++n) {
    const float* r = ctx + ((size_t)lb * N_ + n) * D_;
    __bf16* o = ctxbf + ((size_t)lb * N_ + n) * D_;
#pragma unroll
    for (int j = 0; j < 3; ++j) {
      int d = t + j * 256;
      float v = r[d] * sc[j] + sh[j];
      o[d] = (__bf16)v;
      sum[j] += v;
      mx[j] = fmaxf(mx[j], v);
    }
  }
#pragma unroll
  for (int j = 0; j < 3; ++j) {
    int d = t + j * 256;
    gap[(size_t)lb * D_ + d] = sum[j] * (1.0f / N_);
    gmp[(size_t)lb * D_ + d] = mx[j];
  }
}

// ---------------------------------------------------------------------------
// 6) weight repack: conv weights to bf16 with K=(tap, channel) ordering;
//    convT weights to per-(p,q) [C_out, C_in] A-matrices.
// ---------------------------------------------------------------------------
__global__ void prepw_kernel(const float* __restrict__ lw1, const float* __restrict__ lwt1,
                             const float* __restrict__ lw2, const float* __restrict__ lwt2,
                             __bf16* __restrict__ w1bf, __bf16* __restrict__ wt1A,
                             __bf16* __restrict__ w2bf, __bf16* __restrict__ wt2A) {
  const int N1 = L_ * H_ * 9 * D_;
  const int N2 = L_ * 4 * H_ * H_;
  const int N3 = L_ * MID_ * 9 * H_;
  const int N4 = L_ * 4 * MID_ * MID_;
  int i = blockIdx.x * 256 + threadIdx.x;
  if (i >= N1 + N2 + N3 + N4) return;
  if (i < N1) {
    int l = i / (H_ * 9 * D_); int r = i % (H_ * 9 * D_);
    int h = r / (9 * D_);      int k = r % (9 * D_);
    int t = k / D_;            int d = k % D_;
    w1bf[i] = (__bf16)lw1[((size_t)(l * H_ + h) * D_ + d) * 9 + t];
  } else if (i < N1 + N2) {
    int j = i - N1;
    int l = j / (4 * H_ * H_); int r = j % (4 * H_ * H_);
    int pq = r / (H_ * H_);    int rr = r % (H_ * H_);
    int ho = rr / H_;          int ci = rr % H_;
    wt1A[j] = (__bf16)lwt1[((size_t)(l * H_ + ci) * H_ + ho) * 4 + pq];
  } else if (i < N1 + N2 + N3) {
    int j = i - N1 - N2;
    int l = j / (MID_ * 9 * H_); int r = j % (MID_ * 9 * H_);
    int o = r / (9 * H_);        int k = r % (9 * H_);
    int t = k / H_;              int c = k % H_;
    w2bf[j] = (__bf16)lw2[((size_t)(l * MID_ + o) * H_ + c) * 9 + t];
  } else {
    int j = i - N1 - N2 - N3;
    int l = j / (4 * MID_ * MID_); int r = j % (4 * MID_ * MID_);
    int pq = r / (MID_ * MID_);    int rr = r % (MID_ * MID_);
    int o = rr / MID_;             int ci = rr % MID_;
    wt2A[j] = (__bf16)lwt2[((size_t)(l * MID_ + ci) * MID_ + o) * 4 + pq];
  }
}

// ---------------------------------------------------------------------------
// 7) conv3x3 SAME 768->128 on 16x16 as implicit GEMM [128 x 256], K=9*768.
// ---------------------------------------------------------------------------
__global__ void conv1_kernel(const __bf16* __restrict__ w1bf,
                             const __bf16* __restrict__ ctxbf, float* __restrict__ raw) {
  int lb = blockIdx.x;
  int l = lb >> 4;
  int lane = threadIdx.x & 31, wid = threadIdx.x >> 5;
  const __bf16* bsrc = ctxbf + (size_t)lb * N_ * D_;
  int nloc = lane & 15, boff = (lane >> 4) * 16;
  for (int tile = wid; tile < 128; tile += 8) {
    int cb = tile >> 4, pb = tile & 15;
    int ch0 = cb * 16, pos0 = pb * 16;
    int pos = pos0 + nloc;
    int y = pos >> 4, x = pos & 15;
    v8f acc = zero8f();
    for (int t = 0; t < 9; ++t) {
      int dy = t / 3 - 1, dx = t % 3 - 1;
      int yy = y + dy, xx = x + dx;
      bool ok = (yy >= 0) & (yy < 16) & (xx >= 0) & (xx < 16);
      const __bf16* bp = bsrc + (size_t)(yy * 16 + xx) * D_ + boff;
      const __bf16* ap = w1bf + ((size_t)l * H_ + ch0) * (9 * D_) + t * D_;
      for (int k0 = 0; k0 < D_; k0 += 32) {
        v16bf bf_ = ok ? *(const v16bf*)(bp + k0) : zero16bf();
        v16bf af = frag_a_bf16(ap + k0, 9 * D_, lane);
        acc = wmma_bf16(af, bf_, acc);
      }
    }
#pragma unroll
    for (int r = 0; r < 8; ++r) {
      int ch = ch0 + (lane >> 4) * 8 + r;
      raw[((size_t)lb * H_ + ch) * N_ + pos] = acc[r];
    }
  }
}

// ---------------------------------------------------------------------------
// 8) generic BN stats (per (l,ch) over B*S) -> scale/shift
// ---------------------------------------------------------------------------
__global__ void bnstats_kernel(const float* __restrict__ raw, const float* __restrict__ g,
                               const float* __restrict__ be, float* __restrict__ scale,
                               float* __restrict__ shift, int CH, int S) {
  __shared__ float s1[256], s2[256];
  int ch = blockIdx.x % CH;
  int l = blockIdx.x / CH;
  int t = threadIdx.x;
  int tot = B_ * S;
  float a = 0.f, b2 = 0.f;
  for (int i = t; i < tot; i += 256) {
    int b = i / S, sp = i % S;
    float v = raw[((size_t)(l * B_ + b) * CH + ch) * S + sp];
    a += v; b2 += v * v;
  }
  s1[t] = a; s2[t] = b2; __syncthreads();
  for (int k = 128; k > 0; k >>= 1) {
    if (t < k) { s1[t] += s1[t + k]; s2[t] += s2[t + k]; }
    __syncthreads();
  }
  if (t == 0) {
    float mean = s1[0] / (float)tot;
    float var = s2[0] / (float)tot - mean * mean;
    float sc = g[l * CH + ch] * rsqrtf(var + 1e-5f);
    scale[l * CH + ch] = sc;
    shift[l * CH + ch] = be[l * CH + ch] - mean * sc;
  }
}

// 9) BN apply + ReLU, transpose [ch][pos] -> bf16 [pos][ch]
__global__ void bnapply_kernel(const float* __restrict__ raw,
                               const float* __restrict__ scale,
                               const float* __restrict__ shift,
                               __bf16* __restrict__ outbf, int CH, int S) {
  int lb = blockIdx.x;
  int l = lb >> 4;
  int t = threadIdx.x;
  int tot = CH * S;
  for (int i = t; i < tot; i += 256) {
    int ch = i / S, sp = i % S;
    float v = raw[((size_t)lb * CH + ch) * S + sp] * scale[l * CH + ch] + shift[l * CH + ch];
    outbf[((size_t)lb * S + sp) * CH + ch] = (__bf16)fmaxf(v, 0.f);
  }
}

// ---------------------------------------------------------------------------
// 10) ConvTranspose2d k=2 s=2 as 4 GEMMs [CH x CH] @ [CH x GIN^2], scatter 2x up.
// ---------------------------------------------------------------------------
__global__ void convT_kernel(const __bf16* __restrict__ A, const float* __restrict__ bias,
                             const __bf16* __restrict__ X, __bf16* __restrict__ Y, int CH,
                             int GIN) {
  int lb = blockIdx.x;
  int l = lb >> 4;
  int lane = threadIdx.x & 31, wid = threadIdx.x >> 5;
  int SIN = GIN * GIN;
  int cbN = CH / 16, pbN = SIN / 16;
  int tilesPerPQ = cbN * pbN;
  int totTiles = tilesPerPQ * 4;
  int nloc = lane & 15, boff = (lane >> 4) * 16;
  int GOUT = GIN * 2;
  for (int it = wid; it < totTiles; it += 8) {
    int pq = it / tilesPerPQ;
    int t2 = it % tilesPerPQ;
    int cb = t2 / pbN, pb = t2 % pbN;
    int ch0 = cb * 16, pos0 = pb * 16;
    int pos = pos0 + nloc;
    const __bf16* Ab = A + (((size_t)l * 4 + pq) * CH + ch0) * CH;
    const __bf16* Xb = X + ((size_t)lb * SIN + pos) * CH + boff;
    v8f acc = zero8f();
    for (int k0 = 0; k0 < CH; k0 += 32) {
      v16bf af = frag_a_bf16(Ab + k0, CH, lane);
      v16bf bf_ = *(const v16bf*)(Xb + k0);
      acc = wmma_bf16(af, bf_, acc);
    }
    int p = pq >> 1, q = pq & 1;
    int y = pos / GIN, x = pos % GIN;
    int opos = (2 * y + p) * GOUT + (2 * x + q);
#pragma unroll
    for (int r = 0; r < 8; ++r) {
      int ch = ch0 + (lane >> 4) * 8 + r;
      Y[((size_t)lb * (GOUT * GOUT) + opos) * CH + ch] = (__bf16)(acc[r] + bias[l * CH + ch]);
    }
  }
}

// ---------------------------------------------------------------------------
// 11) conv3x3 SAME 128->64 on 32x32 as implicit GEMM [64 x 1024], K=9*128.
// ---------------------------------------------------------------------------
__global__ void conv2_kernel(const __bf16* __restrict__ w2bf, const __bf16* __restrict__ X,
                             float* __restrict__ raw) {
  int lb = blockIdx.x;
  int l = lb >> 4;
  int lane = threadIdx.x & 31, wid = threadIdx.x >> 5;
  int nloc = lane & 15, boff = (lane >> 4) * 16;
  for (int tile = wid; tile < 256; tile += 8) {
    int cb = tile >> 6, pb = tile & 63;
    int ch0 = cb * 16, pos0 = pb * 16;
    int pos = pos0 + nloc;
    int y = pos >> 5, x = pos & 31;
    v8f acc = zero8f();
    for (int t = 0; t < 9; ++t) {
      int dy = t / 3 - 1, dx = t % 3 - 1;
      int yy = y + dy, xx = x + dx;
      bool ok = (yy >= 0) & (yy < 32) & (xx >= 0) & (xx < 32);
      const __bf16* bp = X + ((size_t)lb * 1024 + yy * 32 + xx) * H_ + boff;
      const __bf16* ap = w2bf + ((size_t)l * MID_ + ch0) * (9 * H_) + t * H_;
      for (int k0 = 0; k0 < H_; k0 += 32) {
        v16bf bf_ = ok ? *(const v16bf*)(bp + k0) : zero16bf();
        v16bf af = frag_a_bf16(ap + k0, 9 * H_, lane);
        acc = wmma_bf16(af, bf_, acc);
      }
    }
#pragma unroll
    for (int r = 0; r < 8; ++r) {
      int ch = ch0 + (lane >> 4) * 8 + r;
      raw[((size_t)lb * MID_ + ch) * 1024 + pos] = acc[r];
    }
  }
}

// 12) conv1x1 64->2 on 64x64
__global__ void head_conv_kernel(const __bf16* __restrict__ Y2, const float* __restrict__ w3,
                                 const float* __restrict__ b3, float* __restrict__ raw3) {
  int blk = blockIdx.x;
  int lb = blk >> 4;
  int l = lb >> 4;
  int pos = (blk & 15) * 256 + threadIdx.x;
  const __bf16* v = Y2 + ((size_t)lb * 4096 + pos) * MID_;
  float s0 = b3[l * 2], s1 = b3[l * 2 + 1];
  for (int c = 0; c < MID_; ++c) {
    float x = (float)v[c];
    s0 += x * w3[(l * 2 + 0) * MID_ + c];
    s1 += x * w3[(l * 2 + 1) * MID_ + c];
  }
  raw3[((size_t)lb * 2 + 0) * 4096 + pos] = s0;
  raw3[((size_t)lb * 2 + 1) * 4096 + pos] = s1;
}

// 13) 4x4 adaptive pool + patch logits/scores
__global__ void pool_kernel(const float* __restrict__ raw3, float* __restrict__ out,
                            size_t O_PS, size_t O_PL) {
  int lb = blockIdx.x;
  int p = threadIdx.x;
  int gy = p >> 4, gx = p & 15;
  float s0 = 0.f, s1 = 0.f;
  for (int i = 0; i < 4; ++i)
    for (int j = 0; j < 4; ++j) {
      int o = (4 * gy + i) * 64 + 4 * gx + j;
      s0 += raw3[((size_t)lb * 2 + 0) * 4096 + o];
      s1 += raw3[((size_t)lb * 2 + 1) * 4096 + o];
    }
  s0 *= (1.f / 16.f);
  s1 *= (1.f / 16.f);
  float pl = s1 - s0;
  out[O_PL + (size_t)lb * 256 + p] = pl;
  out[O_PS + (size_t)lb * 256 + p] = 1.f / (1.f + expf(-pl));
}

// 14) bilinear resize 64 -> 240 (half-pixel centers)
__global__ void resize_kernel(const float* __restrict__ raw3, float* __restrict__ out,
                              size_t O_LL) {
  size_t i = (size_t)blockIdx.x * 256 + threadIdx.x;
  size_t tot = (size_t)L_ * B_ * 2 * IMG_ * IMG_;
  if (i >= tot) return;
  int ox = (int)(i % IMG_);
  int oy = (int)((i / IMG_) % IMG_);
  size_t lbc = i / ((size_t)IMG_ * IMG_);
  const float* src = raw3 + lbc * 4096;
  const float sc = 64.0f / 240.0f;
  float fy = (oy + 0.5f) * sc - 0.5f;
  float fx = (ox + 0.5f) * sc - 0.5f;
  int y0 = (int)floorf(fy), x0 = (int)floorf(fx);
  float wy = fy - y0, wx = fx - x0;
  int y0c = min(max(y0, 0), 63), y1c = min(max(y0 + 1, 0), 63);
  int x0c = min(max(x0, 0), 63), x1c = min(max(x0 + 1, 0), 63);
  float v00 = src[y0c * 64 + x0c], v01 = src[y0c * 64 + x1c];
  float v10 = src[y1c * 64 + x0c], v11 = src[y1c * 64 + x1c];
  out[O_LL + i] = (1.f - wy) * ((1.f - wx) * v00 + wx * v01) +
                  wy * ((1.f - wx) * v10 + wx * v11);
}

// 15) global head: softmax pool weights, 768->128 matvec, LayerNorm, ReLU, 128->2
__global__ void ghead_kernel(const float* __restrict__ gap, const float* __restrict__ gmp,
                             const float* __restrict__ plog, const float* __restrict__ mw1,
                             const float* __restrict__ lng, const float* __restrict__ lnb,
                             const float* __restrict__ mw2, const float* __restrict__ mb2,
                             float* __restrict__ out, size_t O_GL, size_t O_PW) {
  __shared__ float pooled[D_];
  __shared__ float hbuf[H_];
  __shared__ float red[H_];
  __shared__ float l2buf[2];
  int lb = blockIdx.x;
  int l = lb >> 4;
  int b = lb & 15;
  int t = threadIdx.x;
  float p0 = plog[l * 2], p1 = plog[l * 2 + 1];
  float mm = fmaxf(p0, p1);
  float e0 = expf(p0 - mm), e1 = expf(p1 - mm);
  float pw0 = e0 / (e0 + e1), pw1 = e1 / (e0 + e1);
  for (int i = t; i < D_; i += H_)
    pooled[i] = pw0 * gap[(size_t)lb * D_ + i] + pw1 * gmp[(size_t)lb * D_ + i];
  __syncthreads();
  const float* wrow = mw1 + ((size_t)l * H_ + t) * D_;
  float acc = 0.f;
  for (int d = 0; d < D_; ++d) acc += pooled[d] * wrow[d];
  red[t] = acc; __syncthreads();
  for (int k = 64; k > 0; k >>= 1) { if (t < k) red[t] += red[t + k]; __syncthreads(); }
  float mean = red[0] * (1.0f / H_);
  __syncthreads();
  float dv = acc - mean;
  red[t] = dv * dv; __syncthreads();
  for (int k = 64; k > 0; k >>= 1) { if (t < k) red[t] += red[t + k]; __syncthreads(); }
  float var = red[0] * (1.0f / H_);
  float h = lng[l * H_ + t] * (acc - mean) * rsqrtf(var + 1e-5f) + lnb[l * H_ + t];
  hbuf[t] = fmaxf(h, 0.f);
  __syncthreads();
  if (t < 2) {
    float s = mb2[l * 2 + t];
    const float* w2r = mw2 + ((size_t)l * 2 + t) * H_;
    for (int c = 0; c < H_; ++c) s += hbuf[c] * w2r[c];
    l2buf[t] = s;
  }
  __syncthreads();
  if (t == 0) {
    out[O_GL + lb] = l2buf[1] - l2buf[0];
    if (b == 0) { out[O_PW + l * 2] = pw0; out[O_PW + l * 2 + 1] = pw1; }
  }
}

// ---------------------------------------------------------------------------
extern "C" void kernel_launch(void* const* d_in, const int* in_sizes, int n_in,
                              void* d_out, int out_size, void* d_ws, size_t ws_size,
                              hipStream_t stream) {
  (void)in_sizes; (void)n_in; (void)out_size; (void)ws_size;
  const float* query  = (const float*)d_in[0];
  const float* prompt = (const float*)d_in[1];
  const float* sbg    = (const float*)d_in[2];
  const float* sbb    = (const float*)d_in[3];
  const float* lw1    = (const float*)d_in[4];
  const float* lbn1g  = (const float*)d_in[5];
  const float* lbn1b  = (const float*)d_in[6];
  const float* lwt1   = (const float*)d_in[7];
  const float* lbt1   = (const float*)d_in[8];
  const float* lw2    = (const float*)d_in[9];
  const float* lbn2g  = (const float*)d_in[10];
  const float* lbn2b  = (const float*)d_in[11];
  const float* lwt2   = (const float*)d_in[12];
  const float* lbt2   = (const float*)d_in[13];
  const float* lw3    = (const float*)d_in[14];
  const float* lb3    = (const float*)d_in[15];
  const float* plog   = (const float*)d_in[16];
  const float* mw1    = (const float*)d_in[17];
  const float* mlng   = (const float*)d_in[18];
  const float* mlnb   = (const float*)d_in[19];
  const float* mw2    = (const float*)d_in[20];
  const float* mb2    = (const float*)d_in[21];
  float* out = (float*)d_out;

  char* base = (char*)d_ws;
  size_t off = 0;
  auto carve = [&](size_t bytes) -> void* {
    off = (off + 255) & ~(size_t)255;
    void* p = base + off;
    off += bytes;
    return p;
  };
  const size_t QTOT = (size_t)L_ * B_ * N_ * D_;
  float*  qn     = (float*) carve(QTOT * 4);
  float*  prnorm = (float*) carve((size_t)L_ * B_ * M_ * 4);
  float*  maxcos = (float*) carve((size_t)L_ * B_ * N_ * 4);
  int*    idxbuf = (int*)   carve((size_t)L_ * B_ * N_ * 4);
  float*  ctx    = (float*) carve(QTOT * 4);
  float*  bnsc0  = (float*) carve((size_t)L_ * D_ * 4);
  float*  bnsh0  = (float*) carve((size_t)L_ * D_ * 4);
  __bf16* ctxbf  = (__bf16*)carve(QTOT * 2);
  float*  gap    = (float*) carve((size_t)L_ * B_ * D_ * 4);
  float*  gmp    = (float*) carve((size_t)L_ * B_ * D_ * 4);
  __bf16* w1bf   = (__bf16*)carve((size_t)L_ * H_ * 9 * D_ * 2);
  __bf16* wt1A   = (__bf16*)carve((size_t)L_ * 4 * H_ * H_ * 2);
  __bf16* w2bf   = (__bf16*)carve((size_t)L_ * MID_ * 9 * H_ * 2);
  __bf16* wt2A   = (__bf16*)carve((size_t)L_ * 4 * MID_ * MID_ * 2);
  float*  c1raw  = (float*) carve((size_t)L_ * B_ * H_ * N_ * 4);
  float*  bnsc1  = (float*) carve((size_t)L_ * H_ * 4);
  float*  bnsh1  = (float*) carve((size_t)L_ * H_ * 4);
  __bf16* x1bf   = (__bf16*)carve((size_t)L_ * B_ * N_ * H_ * 2);
  __bf16* y1bf   = (__bf16*)carve((size_t)L_ * B_ * 1024 * H_ * 2);
  float*  c2raw  = (float*) carve((size_t)L_ * B_ * MID_ * 1024 * 4);
  float*  bnsc2  = (float*) carve((size_t)L_ * MID_ * 4);
  float*  bnsh2  = (float*) carve((size_t)L_ * MID_ * 4);
  __bf16* x2bf   = (__bf16*)carve((size_t)L_ * B_ * 1024 * MID_ * 2);
  __bf16* y2bf   = (__bf16*)carve((size_t)L_ * B_ * 4096 * MID_ * 2);
  float*  raw3   = (float*) carve((size_t)L_ * B_ * 2 * 4096 * 4);

  const size_t O_RES = 0;
  const size_t O_PS  = 12288;
  const size_t O_PL  = 24576;
  const size_t O_GL  = 36864;
  const size_t O_LL  = 36912;
  const size_t O_PW  = O_LL + (size_t)L_ * B_ * 2 * IMG_ * IMG_;
  const size_t O_IDX = O_PW + 6;
  const size_t O_AL  = O_IDX + 12288;

  rownorm_kernel<<<L_ * B_ * N_, 256, 0, stream>>>(query, qn, nullptr, L_ * B_ * N_);
  rownorm_kernel<<<L_ * B_ * M_, 256, 0, stream>>>(prompt, nullptr, prnorm, L_ * B_ * M_);
  sim_kernel<<<L_ * B_ * 16, 256, 0, stream>>>(qn, prompt, prnorm, maxcos, idxbuf);
  gather_ctx_kernel<<<L_ * B_ * N_, 256, 0, stream>>>(prompt, qn, maxcos, idxbuf, ctx, out,
                                                      O_RES, O_IDX, O_AL);
  bnstats_ctx_kernel<<<L_ * D_, 256, 0, stream>>>(ctx, sbg, sbb, bnsc0, bnsh0);
  bnapply_ctx_kernel<<<L_ * B_, 256, 0, stream>>>(ctx, bnsc0, bnsh0, ctxbf, gap, gmp);
  {
    int tot = L_ * H_ * 9 * D_ + L_ * 4 * H_ * H_ + L_ * MID_ * 9 * H_ + L_ * 4 * MID_ * MID_;
    prepw_kernel<<<(tot + 255) / 256, 256, 0, stream>>>(lw1, lwt1, lw2, lwt2, w1bf, wt1A,
                                                        w2bf, wt2A);
  }
  conv1_kernel<<<L_ * B_, 256, 0, stream>>>(w1bf, ctxbf, c1raw);
  bnstats_kernel<<<L_ * H_, 256, 0, stream>>>(c1raw, lbn1g, lbn1b, bnsc1, bnsh1, H_, N_);
  bnapply_kernel<<<L_ * B_, 256, 0, stream>>>(c1raw, bnsc1, bnsh1, x1bf, H_, N_);
  convT_kernel<<<L_ * B_, 256, 0, stream>>>(wt1A, lbt1, x1bf, y1bf, H_, G_);
  conv2_kernel<<<L_ * B_, 256, 0, stream>>>(w2bf, y1bf, c2raw);
  bnstats_kernel<<<L_ * MID_, 256, 0, stream>>>(c2raw, lbn2g, lbn2b, bnsc2, bnsh2, MID_, 1024);
  bnapply_kernel<<<L_ * B_, 256, 0, stream>>>(c2raw, bnsc2, bnsh2, x2bf, MID_, 1024);
  convT_kernel<<<L_ * B_, 256, 0, stream>>>(wt2A, lbt2, x2bf, y2bf, MID_, 32);
  head_conv_kernel<<<L_ * B_ * 16, 256, 0, stream>>>(y2bf, lw3, lb3, raw3);
  pool_kernel<<<L_ * B_, 256, 0, stream>>>(raw3, out, O_PS, O_PL);
  {
    size_t tot = (size_t)L_ * B_ * 2 * IMG_ * IMG_;
    resize_kernel<<<(unsigned)((tot + 255) / 256), 256, 0, stream>>>(raw3, out, O_LL);
  }
  ghead_kernel<<<L_ * B_, 128, 0, stream>>>(gap, gmp, plog, mw1, mlng, mlnb, mw2, mb2, out,
                                            O_GL, O_PW);
}
